// inter_domain_encoder_50216757624950
// MI455X (gfx1250) — compile-verified, matrix operand order checked by reading, not run on hardware
//
#include <hip/hip_runtime.h>
#include <math.h>

#define DIM   256
#define HEADS 4
#define CDIM  256
#define HC    (HEADS*CDIM)
#define KDIM  256   // inner dim of every GEMM in this model

typedef __attribute__((ext_vector_type(16))) _Float16 v16h;
typedef __attribute__((ext_vector_type(8)))  _Float16 v8h;
typedef __attribute__((ext_vector_type(4)))  _Float16 v4h;
typedef __attribute__((ext_vector_type(8)))  float    v8f;

// ---------------------------------------------------------------- utilities
__global__ void fill_kernel(float* p, float v, long n) {
  long i = (long)blockIdx.x * blockDim.x + threadIdx.x;
  long stride = (long)gridDim.x * blockDim.x;
  for (; i < n; i += stride) p[i] = v;
}

__global__ void deg_kernel(const int* __restrict__ dst, float* deg, int E) {
  int e = blockIdx.x * blockDim.x + threadIdx.x;
  if (e < E) atomicAdd(&deg[dst[e]], 1.0f);
}

__global__ void dinv_kernel(float* deg, int n) {
  int i = blockIdx.x * blockDim.x + threadIdx.x;
  if (i < n) { float d = deg[i]; deg[i] = d > 0.f ? rsqrtf(d) : 0.f; }
}

__device__ __forceinline__ void atomicMaxFloat(float* addr, float val) {
  unsigned int* ai = (unsigned int*)addr;
  unsigned int old = *ai;
  while (__uint_as_float(old) < val) {
    unsigned int assumed = old;
    old = atomicCAS(ai, assumed, __float_as_uint(val));
    if (old == assumed) break;
  }
}

// ---------------------------------------------------------------- WMMA GEMM
// C[M,Nc] = A[M,256] * B + bias(col); optional transposed B access / C store.
// 256 threads = 8 waves. Block tile 64x64, BK=32, K fully unrolled (16 wmma).
// Requires: Nc % 64 == 0, K == 256 (true for every GEMM in this model).
// Pad rows of A are address-clamped only: they feed C rows >= M which are
// never stored, so no value masking is needed.
__global__ void __launch_bounds__(256)
wmma_gemm_kernel(const float* __restrict__ A, const float* __restrict__ B,
                 const float* __restrict__ bias, float* __restrict__ C,
                 int M, int Nc, int transB, int transC)
{
  __shared__ __align__(16) _Float16 As[64 * 40];  // 64 rows x 32 k (stride 40)
  __shared__ __align__(16) _Float16 Bs[64 * 40];  // 64 cols x 32 k (stride 40)

  const int tid  = threadIdx.x;
  const int wave = tid >> 5;
  const int lane = tid & 31;
  const int wr   = wave & 3;       // row tile 0..3
  const int wcg  = wave >> 2;      // col group 0..1 (32 cols each)
  const int half = lane >> 4;      // K-half selector per ISA 16-bit A/B layout
  const int lr   = lane & 15;
  const int rowBase = blockIdx.x * 64;
  const int colBase = blockIdx.y * 64;

  // vectorized cooperative-load coordinates: float4 per access
  const int ldr  = tid >> 3;        // 0..31 (row/col base, +32 second pass)
  const int kk4  = (tid & 7) * 4;   // k offset 0,4,...,28

  v8f acc0 = {};
  v8f acc1 = {};

  #pragma unroll
  for (int k0 = 0; k0 < KDIM; k0 += 32) {
    // ---- A tile: 64 rows x 32 k, two float4 loads per thread ----
    #pragma unroll
    for (int i = 0; i < 2; i++) {
      int r  = ldr + i * 32;
      int gr = rowBase + r;
      int grc = gr < M ? gr : (M - 1);            // clamp address only
      float4 av = *(const float4*)&A[(size_t)grc * KDIM + k0 + kk4];
      v4h hv;
      hv[0] = (_Float16)av.x; hv[1] = (_Float16)av.y;
      hv[2] = (_Float16)av.z; hv[3] = (_Float16)av.w;
      *(v4h*)&As[r * 40 + kk4] = hv;              // ds_store_b64
    }
    // ---- B tile: 64 cols x 32 k, stored [col][k] in LDS ----
    if (transB) {
      #pragma unroll
      for (int i = 0; i < 2; i++) {
        int n = ldr + i * 32;                     // col (always < Nc)
        float4 bv = *(const float4*)&B[(size_t)(colBase + n) * KDIM + k0 + kk4];
        v4h hv;
        hv[0] = (_Float16)bv.x; hv[1] = (_Float16)bv.y;
        hv[2] = (_Float16)bv.z; hv[3] = (_Float16)bv.w;
        *(v4h*)&Bs[n * 40 + kk4] = hv;
      }
    } else {
      int kk = tid >> 3;                          // 0..31 k row of B
      int cb = (tid & 7) * 4;                     // col base 0,4,...,28
      #pragma unroll
      for (int i = 0; i < 2; i++) {
        int c = cb + i * 32;
        float4 bv = *(const float4*)&B[(size_t)(k0 + kk) * Nc + colBase + c];
        Bs[(c + 0) * 40 + kk] = (_Float16)bv.x;   // transpose into LDS
        Bs[(c + 1) * 40 + kk] = (_Float16)bv.y;
        Bs[(c + 2) * 40 + kk] = (_Float16)bv.z;
        Bs[(c + 3) * 40 + kk] = (_Float16)bv.w;
      }
    }
    // next-tile prefetch of A rows (one line = 32 f32 = 128B per row)
    if (k0 + 32 < KDIM && tid < 64) {
      int gr = rowBase + tid;
      int grc = gr < M ? gr : (M - 1);
      __builtin_prefetch(&A[(size_t)grc * KDIM + k0 + 32], 0, 1);
    }
    __syncthreads();

    // ---- fragments: K = j + 8*half (j<8), 16 + (j-8) + 8*half (j>=8) ----
    const int arow  = wr * 16 + lr;
    const int bcol0 = wcg * 32 + lr;
    const int bcol1 = wcg * 32 + 16 + lr;
    v8h a0 = *(const v8h*)&As[arow * 40 + 8 * half];
    v8h a1 = *(const v8h*)&As[arow * 40 + 16 + 8 * half];
    v8h p0 = *(const v8h*)&Bs[bcol0 * 40 + 8 * half];
    v8h p1 = *(const v8h*)&Bs[bcol0 * 40 + 16 + 8 * half];
    v8h q0 = *(const v8h*)&Bs[bcol1 * 40 + 8 * half];
    v8h q1 = *(const v8h*)&Bs[bcol1 * 40 + 16 + 8 * half];
    v16h afrag, bfrag0, bfrag1;
    #pragma unroll
    for (int j = 0; j < 8; j++) {
      afrag[j]  = a0[j]; afrag[j + 8]  = a1[j];
      bfrag0[j] = p0[j]; bfrag0[j + 8] = p1[j];
      bfrag1[j] = q0[j]; bfrag1[j + 8] = q1[j];
    }
    acc0 = __builtin_amdgcn_wmma_f32_16x16x32_f16(
        false, afrag, false, bfrag0, (short)0, acc0, false, false);
    acc1 = __builtin_amdgcn_wmma_f32_16x16x32_f16(
        false, afrag, false, bfrag1, (short)0, acc1, false, false);
    __syncthreads();
  }

  // ---- epilogue: C row = r + 8*half, col = lane&15 (cols always valid) ----
  const int ccol0 = colBase + wcg * 32 + lr;
  const int ccol1 = ccol0 + 16;
  const float bv0 = bias ? bias[ccol0] : 0.f;
  const float bv1 = bias ? bias[ccol1] : 0.f;
  #pragma unroll
  for (int r = 0; r < 8; r++) {
    int row = rowBase + wr * 16 + half * 8 + r;
    if (row < M) {
      float val0 = acc0[r] + bv0;
      float val1 = acc1[r] + bv1;
      if (transC) {
        C[(size_t)ccol0 * M + row] = val0;
        C[(size_t)ccol1 * M + row] = val1;
      } else {
        C[(size_t)row * Nc + ccol0] = val0;
        C[(size_t)row * Nc + ccol1] = val1;
      }
    }
  }
}

// ---------------------------------------------------------------- GCN scatter
__global__ void gcn_scatter_kernel(const int* __restrict__ src, const int* __restrict__ dst,
                                   const float* __restrict__ dinv, const float* __restrict__ h,
                                   float* x, int E) {
  int e = blockIdx.x;
  int d = threadIdx.x;
  int s = src[e], t = dst[e];
  float norm = dinv[s] * dinv[t];
  atomicAdd(&x[(size_t)t * DIM + d], h[(size_t)s * DIM + d] * norm);
}

__global__ void gcn_finalize_kernel(const float* __restrict__ h, const float* __restrict__ dinv,
                                    const float* __restrict__ b_gcn, float* x, int N) {
  long i = (long)blockIdx.x * blockDim.x + threadIdx.x;
  if (i < (long)N * DIM) {
    int n = (int)(i / DIM), d = (int)(i % DIM);
    float di = dinv[n];
    float v = x[i] + h[i] * di * di + b_gcn[d];
    x[i] = fmaxf(v, 0.f);
  }
}

// ---------------------------------------------------------------- attention
__global__ void __launch_bounds__(256)
attn_logits_kernel(const int* __restrict__ src, const int* __restrict__ dst,
                   const float* __restrict__ q, const float* __restrict__ k,
                   float* logits, float* mmax, int E) {
  int gw = (int)(((long)blockIdx.x * blockDim.x + threadIdx.x) >> 5);
  int lane = threadIdx.x & 31;
  if (gw >= E * HEADS) return;
  int e = gw >> 2, hd = gw & 3;
  int s = src[e], t = dst[e];
  const float4* q4 = (const float4*)(q + (size_t)t * HC + hd * CDIM);
  const float4* k4 = (const float4*)(k + (size_t)s * HC + hd * CDIM);
  float4 qa = q4[lane * 2],     ka = k4[lane * 2];
  float4 qb = q4[lane * 2 + 1], kb = k4[lane * 2 + 1];
  float sum = qa.x * ka.x + qa.y * ka.y + qa.z * ka.z + qa.w * ka.w
            + qb.x * kb.x + qb.y * kb.y + qb.z * kb.z + qb.w * kb.w;
  #pragma unroll
  for (int off = 16; off > 0; off >>= 1) sum += __shfl_xor(sum, off, 32);
  sum *= 0.0625f;                                 // 1/sqrt(256)
  if (lane == 0) {
    logits[(size_t)e * HEADS + hd] = sum;
    atomicMaxFloat(&mmax[(size_t)t * HEADS + hd], sum);
  }
}

__global__ void attn_expsum_kernel(const int* __restrict__ dst, float* logits,
                                   const float* __restrict__ mmax, float* denom, int E) {
  int i = blockIdx.x * blockDim.x + threadIdx.x;
  if (i < E * HEADS) {
    int e = i >> 2, hd = i & 3;
    int t = dst[e];
    float ev = __expf(logits[i] - mmax[(size_t)t * HEADS + hd]);
    logits[i] = ev;
    atomicAdd(&denom[(size_t)t * HEADS + hd], ev);
  }
}

__global__ void attn_agg_kernel(const int* __restrict__ src, const int* __restrict__ dst,
                                const float* __restrict__ logits, const float* __restrict__ denom,
                                const float* __restrict__ v, float* outavg, int E) {
  int e = blockIdx.x, d = threadIdx.x;
  int s = src[e], t = dst[e];
  float add = 0.f;
  #pragma unroll
  for (int hd = 0; hd < HEADS; hd++) {
    float ev  = logits[(size_t)e * HEADS + hd];
    float den = denom[(size_t)t * HEADS + hd];
    float alpha = ev / fmaxf(den, 1e-16f);
    add += (0.25f * alpha) * v[(size_t)s * HC + hd * CDIM + d];  // fold head mean
  }
  atomicAdd(&outavg[(size_t)t * DIM + d], add);
}

// ---------------------------------------------------------------- beta gate
__global__ void __launch_bounds__(256)
beta_kernel(const float* __restrict__ outavg, const float* __restrict__ xr,
            const float* __restrict__ Wb, float* tf, int N) {
  __shared__ float red[256];
  int n = blockIdx.x, d = threadIdx.x;
  float o = outavg[(size_t)n * DIM + d];
  float r = xr[(size_t)n * DIM + d];
  red[d] = o * Wb[d] + r * Wb[DIM + d] + (o - r) * Wb[2 * DIM + d];
  __syncthreads();
  for (int off = 128; off > 0; off >>= 1) {
    if (d < off) red[d] += red[d + off];
    __syncthreads();
  }
  float beta = 1.f / (1.f + __expf(-red[0]));
  float val = beta * r + (1.f - beta) * o;
  tf[(size_t)n * DIM + d] = fmaxf(val, 0.f);
}

// ---------------------------------------------------------------- launch
extern "C" void kernel_launch(void* const* d_in, const int* in_sizes, int n_in,
                              void* d_out, int out_size, void* d_ws, size_t ws_size,
                              hipStream_t stream) {
  const float* f_all  = (const float*)d_in[0];
  const int*   eidx   = (const int*)d_in[1];
  const float* W_gcn  = (const float*)d_in[2];
  const float* b_gcn  = (const float*)d_in[3];
  const float* W_q    = (const float*)d_in[4];
  const float* b_q    = (const float*)d_in[5];
  const float* W_k    = (const float*)d_in[6];
  const float* b_k    = (const float*)d_in[7];
  const float* W_v    = (const float*)d_in[8];
  const float* b_v    = (const float*)d_in[9];
  const float* W_skip = (const float*)d_in[10];
  const float* b_skip = (const float*)d_in[11];
  const float* W_beta = (const float*)d_in[12];
  const float* W_cnn  = (const float*)d_in[13];
  const float* b_cnn  = (const float*)d_in[14];

  const int N = in_sizes[0] / DIM;
  const int E = in_sizes[1] / 2;
  const int* src = eidx;
  const int* dst = eidx + E;

  float* ws = (float*)d_ws;
  size_t off = 0;
  float* deg    = ws + off; off += N;                  // degree -> dinv (in place)
  float* h      = ws + off; off += (size_t)N * DIM;
  float* x      = ws + off; off += (size_t)N * DIM;
  float* qb     = ws + off; off += (size_t)N * HC;
  float* kb     = ws + off; off += (size_t)N * HC;
  float* vb     = ws + off; off += (size_t)N * HC;
  float* xr     = ws + off; off += (size_t)N * DIM;
  float* mmax   = ws + off; off += (size_t)N * HEADS;
  float* denom  = ws + off; off += (size_t)N * HEADS;
  float* outavg = ws + off; off += (size_t)N * DIM;
  float* tfb    = ws + off; off += (size_t)N * DIM;
  float* logits = ws + off; off += (size_t)E * HEADS;

  float* outp = (float*)d_out;

  // --- init accumulators (deterministic every call) ---
  fill_kernel<<<1024, 256, 0, stream>>>(deg, 1.0f, (long)N);          // self loop
  fill_kernel<<<1024, 256, 0, stream>>>(mmax, -3.402823466e38f, (long)N * HEADS);
  fill_kernel<<<1024, 256, 0, stream>>>(denom, 0.0f, (long)N * HEADS);
  fill_kernel<<<4096, 256, 0, stream>>>(x, 0.0f, (long)N * DIM);
  fill_kernel<<<4096, 256, 0, stream>>>(outavg, 0.0f, (long)N * DIM);

  // --- GCN normalization ---
  deg_kernel<<<(E + 255) / 256, 256, 0, stream>>>(dst, deg, E);
  dinv_kernel<<<(N + 255) / 256, 256, 0, stream>>>(deg, N);

  // --- h = f_all @ W_gcn (WMMA) ---
  dim3 g1((N + 63) / 64, DIM / 64);
  wmma_gemm_kernel<<<g1, 256, 0, stream>>>(f_all, W_gcn, nullptr, h, N, DIM, 0, 0);

  // --- x = relu(segsum(h[src]*norm -> dst) + self + b_gcn) ---
  gcn_scatter_kernel<<<E, DIM, 0, stream>>>(src, dst, deg, h, x, E);
  gcn_finalize_kernel<<<(int)(((long)N * DIM + 255) / 256), 256, 0, stream>>>(h, deg, b_gcn, x, N);

  // --- q/k/v/skip projections (WMMA) ---
  dim3 gq((N + 63) / 64, HC / 64);
  wmma_gemm_kernel<<<gq, 256, 0, stream>>>(x, W_q, b_q, qb, N, HC, 0, 0);
  wmma_gemm_kernel<<<gq, 256, 0, stream>>>(x, W_k, b_k, kb, N, HC, 0, 0);
  wmma_gemm_kernel<<<gq, 256, 0, stream>>>(x, W_v, b_v, vb, N, HC, 0, 0);
  wmma_gemm_kernel<<<g1, 256, 0, stream>>>(x, W_skip, b_skip, xr, N, DIM, 0, 0);

  // --- edge softmax attention ---
  long totalW = (long)E * HEADS;
  attn_logits_kernel<<<(int)((totalW * 32 + 255) / 256), 256, 0, stream>>>(src, dst, qb, kb, logits, mmax, E);
  attn_expsum_kernel<<<(E * HEADS + 255) / 256, 256, 0, stream>>>(dst, logits, mmax, denom, E);
  attn_agg_kernel<<<E, DIM, 0, stream>>>(src, dst, logits, denom, vb, outavg, E);

  // --- beta gate + relu ---
  beta_kernel<<<N, DIM, 0, stream>>>(outavg, xr, W_beta, tfb, N);

  // --- conv: out[c*N+n] = sum_d W_cnn[c][d]*tf[n][d] + b_cnn[c] (WMMA, B^T, C^T) ---
  wmma_gemm_kernel<<<g1, 256, 0, stream>>>(tfb, W_cnn, b_cnn, outp, N, DIM, 1, 1);

  (void)n_in; (void)out_size; (void)ws_size;
}
